// GATCrossAttentionPretrain_81235011437203
// MI455X (gfx1250) — matrix operand order repaired and checked by reading, not compiled, not run on hardware
//
#include <hip/hip_runtime.h>
#include <hip/hip_bf16.h>
#include <math.h>

#define NNODES 20000
#define NEDGES 320000
#define NGRAPH 512
#define FIN    64
#define HDIM   128
#define EDIM   11
#define OHID   512

typedef __attribute__((ext_vector_type(16))) _Float16 v16h;
typedef __attribute__((ext_vector_type(8)))  _Float16 v8h;
typedef __attribute__((ext_vector_type(8)))  float    v8f;

// ---------------------------------------------------------------------------
// float atomic max via ordered int punning (lowers to global_atomic_max/min)
// ---------------------------------------------------------------------------
__device__ __forceinline__ void atomicMaxFloat(float* addr, float val) {
    int iv = __float_as_int(val);
    if (iv >= 0) atomicMax((int*)addr, iv);
    else         atomicMin((unsigned int*)addr, (unsigned int)iv);
}

// ---------------------------------------------------------------------------
// WMMA GEMM: C[M,N] = A[M,K] @ B[K,N] (+bias) (+relu), f32 in/out,
// f16 operands staged in LDS, f32 accum via v_wmma_f32_16x16x32_f16.
// Block: 256 threads = 8 waves -> 64x64 tile; each wave owns a 16x32 slab
// (two 16x16 WMMA subtiles sharing one A fragment).
// REQUIRES: K % 4 == 0 and N % 4 == 0 (true for all call sites) so every
// float4 staging chunk is fully in- or fully out-of-bounds.
// LDS: A row-major [64][40]; B k-contiguous (transposed) [64][40] so both
// fragments are two contiguous 16B runs per lane -> ds_load_b128.
// ---------------------------------------------------------------------------
#define TM 64
#define TN 64
#define TK 32
#define LDP (TK + 8)

__global__ __launch_bounds__(256) void gemm_wmma(
    const float* __restrict__ A, const float* __restrict__ B,
    const float* __restrict__ bias, float* __restrict__ C,
    int M, int N, int K, int relu)
{
    __shared__ _Float16 As [TM][LDP];
    __shared__ _Float16 Bst[TN][LDP];   // Bst[n][k]

    const int tid  = threadIdx.x;
    const int wave = tid >> 5;
    const int lane = tid & 31;
    const int wm   = wave >> 1;   // 0..3 : 16-row tile
    const int wn   = wave & 1;    // 0..1 : 32-col half
    const int m0   = blockIdx.y * TM;
    const int n0   = blockIdx.x * TN;
    const int am   = lane & 15;
    const int kh   = lane >> 4;

    v8f acc[2] = {};

    for (int k0 = 0; k0 < K; k0 += TK) {
        // ---- stage A tile (64x32): 512 float4 chunks, 2 per thread ----
        #pragma unroll
        for (int i = tid; i < (TM * TK) / 4; i += 256) {
            int r = i >> 3, c = (i & 7) << 2;
            int gm = m0 + r, gk = k0 + c;
            float4 v = {0.f, 0.f, 0.f, 0.f};
            if (gm < M && gk < K)  // chunk fully in-bounds (K % 4 == 0)
                v = *reinterpret_cast<const float4*>(&A[(size_t)gm * K + gk]);
            As[r][c + 0] = (_Float16)v.x;
            As[r][c + 1] = (_Float16)v.y;
            As[r][c + 2] = (_Float16)v.z;
            As[r][c + 3] = (_Float16)v.w;
        }
        // ---- stage B tile (32x64) transposed: 512 float4 chunks ----
        #pragma unroll
        for (int i = tid; i < (TK * TN) / 4; i += 256) {
            int r = i >> 4, c = (i & 15) << 2;    // r = k row, c = n col
            int gk = k0 + r, gn = n0 + c;
            float4 v = {0.f, 0.f, 0.f, 0.f};
            if (gk < K && gn < N)  // chunk fully in-bounds (N % 4 == 0)
                v = *reinterpret_cast<const float4*>(&B[(size_t)gk * N + gn]);
            Bst[c + 0][r] = (_Float16)v.x;
            Bst[c + 1][r] = (_Float16)v.y;
            Bst[c + 2][r] = (_Float16)v.z;
            Bst[c + 3][r] = (_Float16)v.w;
        }
        __syncthreads();

        // ---- A fragment: lane(m=am, khalf=kh) = two contiguous 16B runs ----
        const _Float16* arow = &As[wm * 16 + am][0];
        v8h a0 = *reinterpret_cast<const v8h*>(arow + kh * 8);
        v8h a1 = *reinterpret_cast<const v8h*>(arow + 16 + kh * 8);
        v16h af = __builtin_shufflevector(a0, a1, 0, 1, 2, 3, 4, 5, 6, 7,
                                          8, 9, 10, 11, 12, 13, 14, 15);
        // ---- two B fragments + WMMAs ----
        #pragma unroll
        for (int j = 0; j < 2; ++j) {
            const _Float16* bcol = &Bst[wn * 32 + j * 16 + am][0];
            v8h b0 = *reinterpret_cast<const v8h*>(bcol + kh * 8);
            v8h b1 = *reinterpret_cast<const v8h*>(bcol + 16 + kh * 8);
            v16h bf = __builtin_shufflevector(b0, b1, 0, 1, 2, 3, 4, 5, 6, 7,
                                              8, 9, 10, 11, 12, 13, 14, 15);
            acc[j] = __builtin_amdgcn_wmma_f32_16x16x32_f16(
                false, af, false, bf, (short)0, acc[j], false, false);
        }
        __syncthreads();
    }

    // ---- store: C/D layout lane n = am, row m = r + 8*kh ----
    if (m0 + TM <= M) {
        // fast path: full tile (N is always a multiple of TN here)
        #pragma unroll
        for (int j = 0; j < 2; ++j) {
            int cn = n0 + wn * 32 + j * 16 + am;
            float bv = bias ? bias[cn] : 0.0f;
            #pragma unroll
            for (int r = 0; r < 8; ++r) {
                int cm = m0 + wm * 16 + r + 8 * kh;
                float v = acc[j][r] + bv;
                if (relu) v = fmaxf(v, 0.0f);
                C[(size_t)cm * N + cn] = v;
            }
        }
    } else {
        #pragma unroll
        for (int j = 0; j < 2; ++j) {
            int cn = n0 + wn * 32 + j * 16 + am;
            float bv = bias ? bias[cn] : 0.0f;
            #pragma unroll
            for (int r = 0; r < 8; ++r) {
                int cm = m0 + wm * 16 + r + 8 * kh;
                if (cm < M && cn < N) {
                    float v = acc[j][r] + bv;
                    if (relu) v = fmaxf(v, 0.0f);
                    C[(size_t)cm * N + cn] = v;
                }
            }
        }
    }
}

// ---------------------------------------------------------------------------
// hs[i] = h[i,:]·a_s ; hd[i] = h[i,:]·a_d   (one wave per row)
// ---------------------------------------------------------------------------
__global__ __launch_bounds__(256) void rowdot2(
    const float* __restrict__ h, const float* __restrict__ av,
    const float* __restrict__ dv, float* __restrict__ hs,
    float* __restrict__ hd, int M, int K)
{
    int row  = blockIdx.x * 8 + (threadIdx.x >> 5);
    int lane = threadIdx.x & 31;
    if (row >= M) return;
    float s = 0.0f, d = 0.0f;
    for (int k = lane; k < K; k += 32) {
        float v = h[(size_t)row * K + k];
        s += v * av[k];
        d += v * dv[k];
    }
    for (int off = 16; off; off >>= 1) {
        s += __shfl_down(s, off, 32);
        d += __shfl_down(d, off, 32);
    }
    if (lane == 0) { hs[row] = s; hd[row] = d; }
}

// out[r] = We[r,:]·ae  (r < rows), one wave per row
__global__ void small_matvec(const float* __restrict__ We,
                             const float* __restrict__ ae,
                             float* __restrict__ out, int rows, int K)
{
    int r = blockIdx.x, lane = threadIdx.x;
    if (r >= rows) return;
    float s = 0.0f;
    for (int k = lane; k < K; k += 32) s += We[(size_t)r * K + k] * ae[k];
    for (int off = 16; off; off >>= 1) s += __shfl_down(s, off, 32);
    if (lane == 0) out[r] = s;
}

// broadcast bias into out rows; init m=-inf, den=0
__global__ void init_layer(float* __restrict__ out, const float* __restrict__ b,
                           float* __restrict__ m, float* __restrict__ den, int M, int H)
{
    int i = blockIdx.x * blockDim.x + threadIdx.x;
    if (i < M * H) out[i] = b[i & (H - 1)];
    if (i < M) { m[i] = -INFINITY; den[i] = 0.0f; }
}

__global__ void fill_val(float* __restrict__ p, float v, int n) {
    int i = blockIdx.x * blockDim.x + threadIdx.x;
    if (i < n) p[i] = v;
}

// per-edge logits: leaky_relu(hs[src] + hd[dst] + edge_attr·weae) ; atomicMax m[dst]
__global__ void edge_logits(const int* __restrict__ src, const int* __restrict__ dst,
                            const float* __restrict__ ea, const float* __restrict__ weae,
                            const float* __restrict__ hs, const float* __restrict__ hd,
                            float* __restrict__ lg, float* __restrict__ m, int E)
{
    int e = blockIdx.x * blockDim.x + threadIdx.x;
    if (e >= E) return;
    int d = dst[e];
    float s = hs[src[e]] + hd[d];
    #pragma unroll
    for (int j = 0; j < EDIM; ++j) s += ea[(size_t)e * EDIM + j] * weae[j];
    s = (s > 0.0f) ? s : 0.2f * s;
    lg[e] = s;
    atomicMaxFloat(&m[d], s);
}

// exv = exp(lg - m[seg]); den[seg] += exv
__global__ void seg_exp(const int* __restrict__ seg, const float* __restrict__ lg,
                        const float* __restrict__ m, float* __restrict__ exv,
                        float* __restrict__ den, int n)
{
    int i = blockIdx.x * blockDim.x + threadIdx.x;
    if (i >= n) return;
    int s = seg[i];
    float v = expf(lg[i] - m[s]);
    exv[i] = v;
    atomicAdd(&den[s], v);
}

// out[dst[e]] += (ex[e]/den) * h[src[e]]  (one wave per edge)
__global__ __launch_bounds__(256) void edge_scatter(
    const int* __restrict__ src, const int* __restrict__ dst,
    const float* __restrict__ exv, const float* __restrict__ den,
    const float* __restrict__ h, float* __restrict__ out, int E, int H)
{
    int e = blockIdx.x * 8 + (threadIdx.x >> 5);
    int lane = threadIdx.x & 31;
    if (e >= E) return;
    int s = src[e], d = dst[e];
    float a = exv[e] / (den[d] + 1e-16f);
    for (int f = lane; f < H; f += 32)
        atomicAdd(&out[(size_t)d * H + f], a * h[(size_t)s * H + f]);
}

// out[seg[i]] += (ex[i]/den) * v[i]  (one wave per row)
__global__ __launch_bounds__(256) void seg_scatter(
    const int* __restrict__ seg, const float* __restrict__ exv,
    const float* __restrict__ den, const float* __restrict__ v,
    float* __restrict__ out, int M, int H)
{
    int i = blockIdx.x * 8 + (threadIdx.x >> 5);
    int lane = threadIdx.x & 31;
    if (i >= M) return;
    int d = seg[i];
    float a = exv[i] / (den[d] + 1e-16f);
    for (int f = lane; f < H; f += 32)
        atomicAdd(&out[(size_t)d * H + f], a * v[(size_t)i * H + f]);
}

// cat[i] = [x | f] rowwise
__global__ void concat_xf(const float* __restrict__ x, const float* __restrict__ f,
                          float* __restrict__ cat, int M, int K1, int K2)
{
    int i = blockIdx.x * blockDim.x + threadIdx.x;
    int KT = K1 + K2;
    if (i >= M * KT) return;
    int r = i / KT, c = i - r * KT;
    cat[i] = (c < K1) ? x[(size_t)r * K1 + c] : f[(size_t)r * K2 + (c - K1)];
}

// scores[i] = q[batch[i]]·k[i] * 1/sqrt(H); atomicMax m[batch[i]]
__global__ __launch_bounds__(256) void attn_scores(
    const float* __restrict__ q, const float* __restrict__ k,
    const int* __restrict__ batch, float* __restrict__ sc,
    float* __restrict__ m, int M, int H)
{
    int row  = blockIdx.x * 8 + (threadIdx.x >> 5);
    int lane = threadIdx.x & 31;
    if (row >= M) return;
    int b = batch[row];
    float s = 0.0f;
    for (int t = lane; t < H; t += 32)
        s += q[(size_t)b * H + t] * k[(size_t)row * H + t];
    for (int off = 16; off; off >>= 1) s += __shfl_down(s, off, 32);
    if (lane == 0) {
        float v = s * 0.088388347648318447f;  // 1/sqrt(128)
        sc[row] = v;
        atomicMaxFloat(&m[b], v);
    }
}

// out[row] = z[row,:]·w + b0  (one wave per row)
__global__ __launch_bounds__(256) void final_dot(
    const float* __restrict__ z, const float* __restrict__ w,
    const float* __restrict__ b, float* __restrict__ out, int M, int K)
{
    int row  = blockIdx.x * 8 + (threadIdx.x >> 5);
    int lane = threadIdx.x & 31;
    if (row >= M) return;
    float s = 0.0f;
    for (int k = lane; k < K; k += 32) s += z[(size_t)row * K + k] * w[k];
    for (int off = 16; off; off >>= 1) s += __shfl_down(s, off, 32);
    if (lane == 0) out[row] = s + b[0];
}

// ---------------------------------------------------------------------------
extern "C" void kernel_launch(void* const* d_in, const int* in_sizes, int n_in,
                              void* d_out, int out_size, void* d_ws, size_t ws_size,
                              hipStream_t stream)
{
    (void)in_sizes; (void)n_in; (void)out_size; (void)ws_size;

    const float* x     = (const float*)d_in[0];
    const int*   eidx  = (const int*)d_in[1];
    const float* eattr = (const float*)d_in[2];
    const int*   batch = (const int*)d_in[3];
    const float* metal = (const float*)d_in[4];
    const float* nlW0  = (const float*)d_in[5];
    const float* nlW   = (const float*)d_in[6];
    const float* nl_as = (const float*)d_in[7];
    const float* nl_ad = (const float*)d_in[8];
    const float* nl_We = (const float*)d_in[9];
    const float* nl_ae = (const float*)d_in[10];
    const float* nl_b  = (const float*)d_in[11];
    const float* glW0  = (const float*)d_in[12];
    const float* glW   = (const float*)d_in[13];
    const float* gl_as = (const float*)d_in[14];
    const float* gl_ad = (const float*)d_in[15];
    const float* gl_We = (const float*)d_in[16];
    const float* gl_ae = (const float*)d_in[17];
    const float* gl_b  = (const float*)d_in[18];
    const float* mW1   = (const float*)d_in[19];
    const float* mb1   = (const float*)d_in[20];
    const float* mW2   = (const float*)d_in[21];
    const float* mb2   = (const float*)d_in[22];
    const float* Wq    = (const float*)d_in[23];
    const float* bq    = (const float*)d_in[24];
    const float* Wk    = (const float*)d_in[25];
    const float* bk    = (const float*)d_in[26];
    const float* Wv    = (const float*)d_in[27];
    const float* bv    = (const float*)d_in[28];
    const float* rW1   = (const float*)d_in[29];
    const float* rb1   = (const float*)d_in[30];
    const float* rW2   = (const float*)d_in[31];
    const float* rb2   = (const float*)d_in[32];
    const float* rW3   = (const float*)d_in[33];
    const float* rb3   = (const float*)d_in[34];
    const float* rW4   = (const float*)d_in[35];
    const float* rb4   = (const float*)d_in[36];
    const float* rW5   = (const float*)d_in[37];
    const float* rb5   = (const float*)d_in[38];

    const int* srcI = eidx;
    const int* dstI = eidx + NEDGES;

    // ---- workspace carving (floats) ----
    float* wsp = (float*)d_ws;
    size_t off = 0;
    auto carve = [&](size_t n) { float* p = wsp + off; off += n; return p; };
    float* bufA   = carve((size_t)NNODES * HDIM);
    float* bufB   = carve((size_t)NNODES * HDIM);
    float* bufH   = carve((size_t)NNODES * HDIM);           // layer h-temp
    float* bufCat = carve((size_t)NNODES * (FIN + HDIM));   // aliased later
    float* kbuf   = carve((size_t)NNODES * HDIM);
    float* vbuf   = carve((size_t)NNODES * HDIM);
    float* hs     = carve(NNODES);
    float* hd     = carve(NNODES);
    float* lg     = carve(NEDGES);
    float* ex     = carve(NEDGES);
    float* mseg   = carve(NNODES);
    float* den    = carve(NNODES);
    float* weae   = carve(16);
    // small buffers aliased into bufCat (free after gat layer 0) and lg/ex
    float* mhid    = bufCat;
    float* membed  = bufCat + 65536;
    float* qbuf    = bufCat + 131072;
    float* attnout = bufCat + 196608;
    float* comb    = bufCat + 262144;
    float* z1      = bufCat + 262144 + 131072;
    float* z2      = z1 + 262144;
    float* sc      = lg;   // node-level scores alias edge scratch
    float* sex     = ex;

    auto gemm = [&](const float* A, const float* Bm, const float* bias, float* C,
                    int M, int Nn, int K, int relu) {
        dim3 g((Nn + TN - 1) / TN, (M + TM - 1) / TM);
        gemm_wmma<<<g, 256, 0, stream>>>(A, Bm, bias, C, M, Nn, K, relu);
    };

    const int EB = (NEDGES + 255) / 256;
    const int EW = (NEDGES + 7) / 8;
    const int NW = (NNODES + 7) / 8;

    auto run_layer = [&](const float* xin, int K, const float* W,
                         const float* a_s, const float* a_d,
                         const float* We, const float* a_e,
                         const float* b, float* outbuf) {
        gemm(xin, W, nullptr, bufH, NNODES, HDIM, K, 0);
        rowdot2<<<NW, 256, 0, stream>>>(bufH, a_s, a_d, hs, hd, NNODES, HDIM);
        small_matvec<<<EDIM, 32, 0, stream>>>(We, a_e, weae, EDIM, HDIM);
        init_layer<<<((size_t)NNODES * HDIM + 255) / 256, 256, 0, stream>>>(
            outbuf, b, mseg, den, NNODES, HDIM);
        edge_logits<<<EB, 256, 0, stream>>>(srcI, dstI, eattr, weae, hs, hd, lg, mseg, NEDGES);
        seg_exp<<<EB, 256, 0, stream>>>(dstI, lg, mseg, ex, den, NEDGES);
        edge_scatter<<<EW, 256, 0, stream>>>(srcI, dstI, ex, den, bufH, outbuf, NEDGES, HDIM);
    };

    // ---- node_layers stack ----
    run_layer(x,    FIN,  nlW0,                nl_as + 0 * HDIM, nl_ad + 0 * HDIM,
              nl_We + 0 * EDIM * HDIM, nl_ae + 0 * HDIM, nl_b + 0 * HDIM, bufA);
    run_layer(bufA, HDIM, nlW + 0 * HDIM * HDIM, nl_as + 1 * HDIM, nl_ad + 1 * HDIM,
              nl_We + 1 * EDIM * HDIM, nl_ae + 1 * HDIM, nl_b + 1 * HDIM, bufB);
    run_layer(bufB, HDIM, nlW + 1 * HDIM * HDIM, nl_as + 2 * HDIM, nl_ad + 2 * HDIM,
              nl_We + 2 * EDIM * HDIM, nl_ae + 2 * HDIM, nl_b + 2 * HDIM, bufA);
    run_layer(bufA, HDIM, nlW + 2 * HDIM * HDIM, nl_as + 3 * HDIM, nl_ad + 3 * HDIM,
              nl_We + 3 * EDIM * HDIM, nl_ae + 3 * HDIM, nl_b + 3 * HDIM, bufB);

    // ---- concat [x | f] ----
    concat_xf<<<((size_t)NNODES * (FIN + HDIM) + 255) / 256, 256, 0, stream>>>(
        x, bufB, bufCat, NNODES, FIN, HDIM);

    // ---- gat_layers stack ----
    run_layer(bufCat, FIN + HDIM, glW0,          gl_as + 0 * HDIM, gl_ad + 0 * HDIM,
              gl_We + 0 * EDIM * HDIM, gl_ae + 0 * HDIM, gl_b + 0 * HDIM, bufA);
    run_layer(bufA, HDIM, glW + 0 * HDIM * HDIM, gl_as + 1 * HDIM, gl_ad + 1 * HDIM,
              gl_We + 1 * EDIM * HDIM, gl_ae + 1 * HDIM, gl_b + 1 * HDIM, bufB);
    run_layer(bufB, HDIM, glW + 1 * HDIM * HDIM, gl_as + 2 * HDIM, gl_ad + 2 * HDIM,
              gl_We + 2 * EDIM * HDIM, gl_ae + 2 * HDIM, gl_b + 2 * HDIM, bufA);
    run_layer(bufA, HDIM, glW + 2 * HDIM * HDIM, gl_as + 3 * HDIM, gl_ad + 3 * HDIM,
              gl_We + 3 * EDIM * HDIM, gl_ae + 3 * HDIM, gl_b + 3 * HDIM, bufB);
    const float* hfin = bufB;

    // ---- metal MLP + projections ----
    gemm(metal,  mW1, mb1, mhid,   NGRAPH, HDIM, 8,    1);
    gemm(mhid,   mW2, mb2, membed, NGRAPH, HDIM, HDIM, 0);
    gemm(membed, Wq,  bq,  qbuf,   NGRAPH, HDIM, HDIM, 0);
    gemm(hfin,   Wk,  bk,  kbuf,   NNODES, HDIM, HDIM, 0);
    gemm(hfin,   Wv,  bv,  vbuf,   NNODES, HDIM, HDIM, 0);

    // ---- batch-segmented cross attention ----
    fill_val<<<2, 256, 0, stream>>>(mseg, -INFINITY, NGRAPH);
    fill_val<<<2, 256, 0, stream>>>(den, 0.0f, NGRAPH);
    fill_val<<<(NGRAPH * HDIM + 255) / 256, 256, 0, stream>>>(attnout, 0.0f, NGRAPH * HDIM);
    attn_scores<<<NW, 256, 0, stream>>>(qbuf, kbuf, batch, sc, mseg, NNODES, HDIM);
    seg_exp<<<(NNODES + 255) / 256, 256, 0, stream>>>(batch, sc, mseg, sex, den, NNODES);
    seg_scatter<<<NW, 256, 0, stream>>>(batch, sex, den, vbuf, attnout, NNODES, HDIM);

    // ---- combined + regressor ----
    concat_xf<<<(NGRAPH * 2 * HDIM + 255) / 256, 256, 0, stream>>>(
        attnout, membed, comb, NGRAPH, HDIM, HDIM);
    gemm(comb, rW1, rb1, z1, NGRAPH, OHID, 2 * HDIM, 1);
    gemm(z1,   rW2, rb2, z2, NGRAPH, OHID, OHID,     1);
    gemm(z2,   rW3, rb3, z1, NGRAPH, OHID, OHID,     1);
    gemm(z1,   rW4, rb4, z2, NGRAPH, 256,  OHID,     1);
    final_dot<<<(NGRAPH + 7) / 8, 256, 0, stream>>>(z2, rW5, rb5, (float*)d_out, NGRAPH, 256);
}
